// SpatialMouseGNN_25537875542266
// MI455X (gfx1250) — matrix operand order repaired and checked by reading, not compiled, not run on hardware
//
#include <hip/hip_runtime.h>
#include <hip/hip_bf16.h>

typedef _Float16 h16;
typedef __attribute__((ext_vector_type(16))) _Float16 v16h;
typedef __attribute__((ext_vector_type(8)))  _Float16 v8h;
typedef __attribute__((ext_vector_type(8)))  float    v8f;
typedef __attribute__((ext_vector_type(4)))  int      v4i;

#define WMMA_F16(a, b, c) \
  __builtin_amdgcn_wmma_f32_16x16x32_f16(false, (a), false, (b), (short)0, (c), false, false)

#if defined(__has_builtin)
#if __has_builtin(__builtin_amdgcn_global_load_async_to_lds_b128) && \
    __has_builtin(__builtin_amdgcn_s_wait_asynccnt)
#define HAVE_ASYNC_LDS 1
typedef __attribute__((address_space(1))) v4i* g_b128_t;  // global int4*
typedef __attribute__((address_space(3))) v4i* l_b128_t;  // LDS int4*
#endif
#endif

// ---------------------------------------------------------------------------
// WMMA fragment loaders (gfx1250 wave32; A and B both use contiguous k = hi*16 + elem)
// A (16x32 f16, row-major MxK): lane = hi*16 + r; elems 0..7  = K[k0+hi*16 .. +7]
//                                               elems 8..15 = K[k0+hi*16+8 .. +7]
// B (32x16 f16) loaded from pre-transposed W^T [N][K] row-major:
//   lane = hi*16 + n; elems 0..15 = K[k0+hi*16 .. +15]  (one contiguous 32B load)
// C/D (16x16 f32): elem e -> row (e + 8*hi), col (lane & 15)
// ---------------------------------------------------------------------------
__device__ __forceinline__ v16h frag_a_f16(const h16* p, int ld) {
  int lane = threadIdx.x & 31;
  int r = lane & 15, hi = lane >> 4;
  const h16* q = p + (size_t)r * ld + hi * 16;
  v8h lo = *(const v8h*)(q);
  v8h up = *(const v8h*)(q + 8);
  v16h f;
#pragma unroll
  for (int i = 0; i < 8; ++i) { f[i] = lo[i]; f[i + 8] = up[i]; }
  return f;
}

__device__ __forceinline__ v16h frag_b_f16(const h16* __restrict__ p, int ld) {
  int lane = threadIdx.x & 31;
  int n = lane & 15, hi = lane >> 4;
  return *(const v16h*)(p + (size_t)n * ld + hi * 16);
}

// ---------------------------------------------------------------------------
// Weight prep: f32 [K][N] row-major  ->  f16 [N][K] row-major (transpose+cvt)
// ---------------------------------------------------------------------------
__global__ void k_transpose_cvt(const float* __restrict__ src, h16* __restrict__ dst,
                                int K, int N) {
  int idx = blockIdx.x * blockDim.x + threadIdx.x;
  if (idx >= K * N) return;
  int n = idx / K, k = idx - n * K;
  dst[idx] = (h16)src[(size_t)k * N + n];
}

// ---------------------------------------------------------------------------
// Embedding: h0 = relu(x @ emb_w + emb_b), x f32 [N][128], WT f16 [256][128]
// 32 rows per workgroup, 8 waves; wave w owns cols [w*32, w*32+32).
// X tile staged once into LDS (f32->f16) to avoid 8x redundant global reads.
// ---------------------------------------------------------------------------
__global__ __launch_bounds__(256) void k_embed(const float* __restrict__ X,
                                               const h16* __restrict__ WT,
                                               const float* __restrict__ bias,
                                               h16* __restrict__ Hout) {
  __shared__ h16 ldsA[32][128 + 8];
  const int wave = threadIdx.x >> 5;
  const int lane = threadIdx.x & 31;
  const int row0 = blockIdx.x * 32;

  // cooperative stage: 32x128 f32 -> f16 in LDS (16 elems / thread)
  {
    int r = threadIdx.x >> 3;            // 0..31
    int c0 = (threadIdx.x & 7) * 16;     // 0..112
    const float* src = X + (size_t)(row0 + r) * 128 + c0;
#pragma unroll
    for (int u = 0; u < 16; ++u) ldsA[r][c0 + u] = (h16)src[u];
  }
  __syncthreads();

  v8f acc[2][2] = {};
#pragma unroll
  for (int k0 = 0; k0 < 128; k0 += 32) {
    v16h a0 = frag_a_f16(&ldsA[0][0] + k0, 128 + 8);
    v16h a1 = frag_a_f16(&ldsA[16][0] + k0, 128 + 8);
#pragma unroll
    for (int t = 0; t < 2; ++t) {
      v16h b = frag_b_f16(WT + (size_t)(wave * 32 + t * 16) * 128 + k0, 128);
      acc[0][t] = WMMA_F16(a0, b, acc[0][t]);
      acc[1][t] = WMMA_F16(a1, b, acc[1][t]);
    }
  }
  const int hi = lane >> 4, nl = lane & 15;
#pragma unroll
  for (int rt = 0; rt < 2; ++rt)
#pragma unroll
    for (int t = 0; t < 2; ++t) {
      int col = wave * 32 + t * 16 + nl;
      float bb = bias[col];
#pragma unroll
      for (int e = 0; e < 8; ++e) {
        int row = row0 + rt * 16 + e + 8 * hi;
        float v = acc[rt][t][e] + bb;
        Hout[(size_t)row * 256 + col] = (h16)(v > 0.f ? v : 0.f);
      }
    }
}

// ---------------------------------------------------------------------------
// Fused TransformerConv layer (+beta gate + LayerNorm + ReLU)
// Hin f16 [N][256]; WT f16 [1024][256] = [q^T; k^T; v^T; s^T]
// 32 rows (8 frames) per workgroup, 256 threads (8 waves).
// A tile staged into LDS (async global->LDS where available); each weight
// (B) fragment feeds 2 WMMAs; epilogue = two 16-row passes reusing LDS.
// ---------------------------------------------------------------------------
__global__ __launch_bounds__(256) void k_conv(const h16* __restrict__ Hin,
                                              const h16* __restrict__ WT,
                                              const float* __restrict__ qb,
                                              const float* __restrict__ kb,
                                              const float* __restrict__ vb,
                                              const float* __restrict__ sb,
                                              const float* __restrict__ bw,
                                              const float* __restrict__ lng,
                                              const float* __restrict__ lnb,
                                              h16* __restrict__ Hout16,
                                              float* __restrict__ OutF32) {
  __shared__ h16 ldsY[16][1024 + 8];  // qkvs (f16) per row: [q|k|v|x_r]
  __shared__ union {
    h16   A[32][256 + 8];             // staged A tile (phase 1 only)
    float O[16][256 + 4];             // attention output (epilogue only)
  } u;

  const int wave = threadIdx.x >> 5;
  const int lane = threadIdx.x & 31;
  const int row0 = blockIdx.x * 32;
  const int LDA = 256 + 8;

  // ---- Stage 0: cooperative copy of Hin tile (32x256 f16) into LDS ----
  {
#pragma unroll
    for (int i = 0; i < 4; ++i) {
      int id = threadIdx.x + i * 256;   // 1024 16-byte chunks
      int r = id >> 5, c = id & 31;     // 32 chunks per row
      const h16* g = Hin + (size_t)(row0 + r) * 256 + c * 8;
      h16* l = &u.A[r][c * 8];
#ifdef HAVE_ASYNC_LDS
      __builtin_amdgcn_global_load_async_to_lds_b128(
          (g_b128_t)(void*)g, (l_b128_t)(void*)l, 0, 0);
#else
      *(v8h*)l = *(const v8h*)g;
#endif
    }
#ifdef HAVE_ASYNC_LDS
    __builtin_amdgcn_s_wait_asynccnt(0);
#endif
  }
  __syncthreads();

  // ---- Phase 1: Y[32x1024] = A_tile @ [Wq|Wk|Wv|Ws]  (WMMA) ----
  v8f acc[2][8] = {};
  for (int k0 = 0; k0 < 256; k0 += 32) {
    v16h a0 = frag_a_f16(&u.A[0][0] + k0, LDA);
    v16h a1 = frag_a_f16(&u.A[16][0] + k0, LDA);
#pragma unroll
    for (int t = 0; t < 8; ++t) {
      v16h b = frag_b_f16(WT + (size_t)(wave * 128 + t * 16) * 256 + k0, 256);
      acc[0][t] = WMMA_F16(a0, b, acc[0][t]);
      acc[1][t] = WMMA_F16(a1, b, acc[1][t]);
    }
  }

  // ---- Fused epilogue: two 16-row passes sharing the LDS buffers ----
  const int hi = lane >> 4, nl = lane & 15;
  for (int rt = 0; rt < 2; ++rt) {
    __syncthreads();  // phase-1 A reads / previous pass done before LDS reuse
    const int rbase = row0 + rt * 16;

    // store this row tile (+bias) to LDS
#pragma unroll
    for (int t = 0; t < 8; ++t) {
      int col = wave * 128 + t * 16 + nl;
      float bb = (col < 256) ? qb[col]
               : (col < 512) ? kb[col - 256]
               : (col < 768) ? vb[col - 512]
                             : sb[col - 768];
#pragma unroll
      for (int e = 0; e < 8; ++e)
        ldsY[e + 8 * hi][col] = (h16)(acc[rt][t][e] + bb);
    }
    __syncthreads();

    // attention: M=4 nodes/frame, H=4 heads, C=64, no self edges
    {
      int pair = threadIdx.x >> 2;  // 64 (row, head) pairs
      int c4   = threadIdx.x & 3;   // quarter of the 64 channels
      int r  = pair >> 2;           // row in tile (0..15)
      int h  = pair & 3;            // head
      int fr = r >> 2;              // frame in tile
      int i  = r & 3;               // node index
      const h16* qrow = &ldsY[r][h * 64];
      float sc[4];
#pragma unroll
      for (int j = 0; j < 4; ++j) {
        if (j == i) { sc[j] = -1e30f; continue; }
        const h16* krow = &ldsY[fr * 4 + j][256 + h * 64];
        float s = 0.f;
#pragma unroll
        for (int c = 0; c < 64; ++c) s += (float)qrow[c] * (float)krow[c];
        sc[j] = s * 0.125f;  // 1/sqrt(64)
      }
      float mx = fmaxf(fmaxf(sc[0], sc[1]), fmaxf(sc[2], sc[3]));
      float den = 0.f;
#pragma unroll
      for (int j = 0; j < 4; ++j) {
        sc[j] = (j == i) ? 0.f : __expf(sc[j] - mx);
        den += sc[j];
      }
      float inv = 1.f / den;
      for (int c = c4 * 16; c < c4 * 16 + 16; ++c) {
        float o = 0.f;
#pragma unroll
        for (int j = 0; j < 4; ++j)
          o += sc[j] * (float)ldsY[fr * 4 + j][512 + h * 64 + c];
        u.O[r][h * 64 + c] = o * inv;
      }
    }
    __syncthreads();

    // beta gate + blend + LayerNorm + ReLU
    {
      int r   = threadIdx.x >> 4;   // row in tile
      int sub = threadIdx.x & 15;   // 16 lanes per row, 16 cols each
      int cbase = sub * 16;
      float o[16], xr[16];
      float bsum = 0.f;
#pragma unroll
      for (int uu = 0; uu < 16; ++uu) {
        int c = cbase + uu;
        o[uu]  = u.O[r][c];
        xr[uu] = (float)ldsY[r][768 + c];
        bsum += o[uu] * bw[c] + xr[uu] * bw[256 + c] + (o[uu] - xr[uu]) * bw[512 + c];
      }
#pragma unroll
      for (int m = 1; m <= 8; m <<= 1) bsum += __shfl_xor(bsum, m, 32);
      float g = 1.f / (1.f + __expf(-bsum));

      float y[16], s1 = 0.f;
#pragma unroll
      for (int uu = 0; uu < 16; ++uu) { y[uu] = g * xr[uu] + (1.f - g) * o[uu]; s1 += y[uu]; }
#pragma unroll
      for (int m = 1; m <= 8; m <<= 1) s1 += __shfl_xor(s1, m, 32);
      float mu = s1 * (1.f / 256.f);

      float s2 = 0.f;
#pragma unroll
      for (int uu = 0; uu < 16; ++uu) { float d = y[uu] - mu; s2 += d * d; }
#pragma unroll
      for (int m = 1; m <= 8; m <<= 1) s2 += __shfl_xor(s2, m, 32);
      float rstd = rsqrtf(s2 * (1.f / 256.f) + 1e-5f);

      size_t rowg = (size_t)(rbase + r) * 256;
#pragma unroll
      for (int uu = 0; uu < 16; ++uu) {
        int c = cbase + uu;
        float z = (y[uu] - mu) * rstd * lng[c] + lnb[c];
        z = z > 0.f ? z : 0.f;
        if (Hout16) Hout16[rowg + c] = (h16)z;
        if (OutF32) OutF32[rowg + c] = z;
      }
    }
  }
}

// ---------------------------------------------------------------------------
extern "C" void kernel_launch(void* const* d_in, const int* in_sizes, int n_in,
                              void* d_out, int out_size, void* d_ws, size_t ws_size,
                              hipStream_t stream) {
  const float* x     = (const float*)d_in[0];
  const float* emb_w = (const float*)d_in[1];
  const float* emb_b = (const float*)d_in[2];
  const float* q1w = (const float*)d_in[3];  const float* q1b = (const float*)d_in[4];
  const float* k1w = (const float*)d_in[5];  const float* k1b = (const float*)d_in[6];
  const float* v1w = (const float*)d_in[7];  const float* v1b = (const float*)d_in[8];
  const float* s1w = (const float*)d_in[9];  const float* s1b = (const float*)d_in[10];
  const float* b1w = (const float*)d_in[11];
  const float* ln1g = (const float*)d_in[12]; const float* ln1b = (const float*)d_in[13];
  const float* q2w = (const float*)d_in[14]; const float* q2b = (const float*)d_in[15];
  const float* k2w = (const float*)d_in[16]; const float* k2b = (const float*)d_in[17];
  const float* v2w = (const float*)d_in[18]; const float* v2b = (const float*)d_in[19];
  const float* s2w = (const float*)d_in[20]; const float* s2b = (const float*)d_in[21];
  const float* b2w = (const float*)d_in[22];
  const float* ln2g = (const float*)d_in[23]; const float* ln2b = (const float*)d_in[24];

  const int nrows = in_sizes[0] / 128;  // B*T*M = 131072

  char* ws = (char*)d_ws;
  h16* WTe = (h16*)ws;                                  // [256][128]
  h16* WT1 = (h16*)(ws + 65536);                        // [1024][256]
  h16* WT2 = (h16*)(ws + 65536 + 524288);               // [1024][256]
  h16* h0  = (h16*)(ws + 65536 + 2 * 524288);           // [N][256]
  h16* h1  = h0 + (size_t)nrows * 256;                  // [N][256]

  // ---- weight prep (transpose + f16 convert) ----
  k_transpose_cvt<<<(256 * 128 + 255) / 256, 256, 0, stream>>>(emb_w, WTe, 128, 256);
  k_transpose_cvt<<<(256 * 256 + 255) / 256, 256, 0, stream>>>(q1w, WT1 + 0 * 256 * 256, 256, 256);
  k_transpose_cvt<<<(256 * 256 + 255) / 256, 256, 0, stream>>>(k1w, WT1 + 1 * 256 * 256, 256, 256);
  k_transpose_cvt<<<(256 * 256 + 255) / 256, 256, 0, stream>>>(v1w, WT1 + 2 * 256 * 256, 256, 256);
  k_transpose_cvt<<<(256 * 256 + 255) / 256, 256, 0, stream>>>(s1w, WT1 + 3 * 256 * 256, 256, 256);
  k_transpose_cvt<<<(256 * 256 + 255) / 256, 256, 0, stream>>>(q2w, WT2 + 0 * 256 * 256, 256, 256);
  k_transpose_cvt<<<(256 * 256 + 255) / 256, 256, 0, stream>>>(k2w, WT2 + 1 * 256 * 256, 256, 256);
  k_transpose_cvt<<<(256 * 256 + 255) / 256, 256, 0, stream>>>(v2w, WT2 + 2 * 256 * 256, 256, 256);
  k_transpose_cvt<<<(256 * 256 + 255) / 256, 256, 0, stream>>>(s2w, WT2 + 3 * 256 * 256, 256, 256);

  // ---- embedding + ReLU ----
  k_embed<<<nrows / 32, 256, 0, stream>>>(x, WTe, emb_b, h0);

  // ---- conv1 -> h1 (f16), conv2 -> d_out (f32) ----
  k_conv<<<nrows / 32, 256, 0, stream>>>(h0, WT1, q1b, k1b, v1b, s1b, b1w, ln1g, ln1b,
                                         h1, (float*)nullptr);
  k_conv<<<nrows / 32, 256, 0, stream>>>(h1, WT2, q2b, k2b, v2b, s2b, b2w, ln2g, ln2b,
                                         (h16*)nullptr, (float*)d_out);
}
